// LSTM_55886114456235
// MI455X (gfx1250) — matrix operand order repaired and checked by reading, not compiled, not run on hardware
//
#include <hip/hip_runtime.h>

typedef __attribute__((ext_vector_type(16))) _Float16 v16h;
typedef __attribute__((ext_vector_type(8)))  float    v8f;

#define LSTM_H   128
#define LSTM_T   8192
#define NTHREADS 512          // 16 waves (wave32)
#define MT_PER_WAVE 2         // 32 M-tiles of 16 gate-rows / 16 waves

__device__ __forceinline__ float fast_sigmoid(float x) {
    return 1.0f / (1.0f + __expf(-x));
}
__device__ __forceinline__ float fast_tanh(float x) {
    // tanh(x) = 1 - 2/(exp(2x)+1)
    return 1.0f - 2.0f / (__expf(2.0f * x) + 1.0f);
}

__global__ __launch_bounds__(NTHREADS, 1)
void lstm_persistent_kernel(const float* __restrict__ x,
                            const float* __restrict__ W_ih,
                            const float* __restrict__ W_hh,
                            const float* __restrict__ b_ih,
                            const float* __restrict__ b_hh,
                            const float* __restrict__ W_lin,
                            const float* __restrict__ b_lin,
                            const float* __restrict__ h0,
                            const float* __restrict__ c0,
                            float* __restrict__ out)
{
    __shared__ float xs[LSTM_T];                         // 32 KB: whole input sequence
    __shared__ __align__(32) _Float16 hs[LSTM_H];        // current h, f16, B-matrix friendly
    __shared__ float gs[4 * LSTM_H];                     // W_hh @ h result (512 gate pre-acts)
    __shared__ float red[LSTM_H];                        // final reduction

    const int tid  = threadIdx.x;
    const int lane = tid & 31;
    const int wave = tid >> 5;

    // ---- one-time staging -------------------------------------------------
    for (int i = tid; i < LSTM_T; i += NTHREADS) xs[i] = x[i];

    float c_reg = 0.0f, h_reg = 0.0f;
    float wih[4]  = {0.f, 0.f, 0.f, 0.f};
    float bias[4] = {0.f, 0.f, 0.f, 0.f};
    if (tid < LSTM_H) {
        c_reg = c0[tid];
        h_reg = h0[tid];
        hs[tid] = (_Float16)h_reg;
        #pragma unroll
        for (int g = 0; g < 4; ++g) {
            wih[g]  = W_ih[g * LSTM_H + tid];
            bias[g] = b_ih[g * LSTM_H + tid] + b_hh[g * LSTM_H + tid];
        }
    }

    // ---- preload W_hh as f16 WMMA A-fragments into VGPRs ------------------
    // A layout (f16 16x32): lanes 0-15 hold M=lane, K=0-15;
    //                       lanes 16-31 hold M=lane-16, K=16-31.
    v16h a[MT_PER_WAVE][4];
    const int mrow  = lane & 15;
    const int khalf = (lane >> 4) ? 16 : 0;
    #pragma unroll
    for (int mt = 0; mt < MT_PER_WAVE; ++mt) {
        const int row = (wave * MT_PER_WAVE + mt) * 16 + mrow;  // gate row 0..511
        const float* wrow = W_hh + row * LSTM_H;
        #pragma unroll
        for (int kt = 0; kt < 4; ++kt) {
            v16h frag;
            #pragma unroll
            for (int e = 0; e < 16; ++e) {
                const int vg = e >> 1, p = e & 1;
                const int k  = kt * 32 + khalf + 2 * vg + p;
                frag[e] = (_Float16)wrow[k];
            }
            a[mt][kt] = frag;
        }
    }
    __syncthreads();

    // B layout (f16 32x16): lanes 0-15 hold K=0-15, lanes 16-31 hold K=16-31,
    // contiguous in K -> plain 32B-aligned LDS vector loads (h broadcast over N).
    const int kb = (lane >> 4) ? 16 : 0;

    // ---- sequential recurrence -------------------------------------------
    for (int t = 0; t < LSTM_T; ++t) {
        v16h b0 = *(const v16h*)(hs + 0  + kb);
        v16h b1 = *(const v16h*)(hs + 32 + kb);
        v16h b2 = *(const v16h*)(hs + 64 + kb);
        v16h b3 = *(const v16h*)(hs + 96 + kb);

        v8f acc[MT_PER_WAVE];
        #pragma unroll
        for (int mt = 0; mt < MT_PER_WAVE; ++mt) {
            v8f cacc = {};
            cacc = __builtin_amdgcn_wmma_f32_16x16x32_f16(false, a[mt][0], false, b0, (short)0, cacc, false, false);
            cacc = __builtin_amdgcn_wmma_f32_16x16x32_f16(false, a[mt][1], false, b1, (short)0, cacc, false, false);
            cacc = __builtin_amdgcn_wmma_f32_16x16x32_f16(false, a[mt][2], false, b2, (short)0, cacc, false, false);
            cacc = __builtin_amdgcn_wmma_f32_16x16x32_f16(false, a[mt][3], false, b3, (short)0, cacc, false, false);
            acc[mt] = cacc;
        }

        // D column 0: lane 0 has rows v (VGPR v), lane 16 has rows 8+v.
        if ((lane & 15) == 0) {
            const int half = (lane >> 4) * 8;
            #pragma unroll
            for (int mt = 0; mt < MT_PER_WAVE; ++mt) {
                const int base = (wave * MT_PER_WAVE + mt) * 16 + half;
                #pragma unroll
                for (int v = 0; v < 8; ++v) gs[base + v] = acc[mt][v];
            }
        }
        __syncthreads();

        if (tid < LSTM_H) {
            const float xt = xs[t];
            const float gi = gs[tid]              + xt * wih[0] + bias[0];
            const float gf = gs[LSTM_H + tid]     + xt * wih[1] + bias[1];
            const float gg = gs[2 * LSTM_H + tid] + xt * wih[2] + bias[2];
            const float go = gs[3 * LSTM_H + tid] + xt * wih[3] + bias[3];
            const float i = fast_sigmoid(gi);
            const float f = fast_sigmoid(gf);
            const float g = fast_tanh(gg);
            const float o = fast_sigmoid(go);
            c_reg = f * c_reg + i * g;
            h_reg = o * fast_tanh(c_reg);
            hs[tid] = (_Float16)h_reg;
        }
        __syncthreads();
    }

    // ---- final projection: out = h_T . W_lin + b_lin ----------------------
    if (tid < LSTM_H) red[tid] = h_reg * W_lin[tid];
    __syncthreads();
    if (tid == 0) {
        float s = 0.0f;
        for (int i = 0; i < LSTM_H; ++i) s += red[i];
        out[0] = s + b_lin[0];
    }
}

extern "C" void kernel_launch(void* const* d_in, const int* in_sizes, int n_in,
                              void* d_out, int out_size, void* d_ws, size_t ws_size,
                              hipStream_t stream) {
    (void)in_sizes; (void)n_in; (void)d_ws; (void)ws_size; (void)out_size;
    const float* x     = (const float*)d_in[0];
    const float* W_ih  = (const float*)d_in[1];
    const float* W_hh  = (const float*)d_in[2];
    const float* b_ih  = (const float*)d_in[3];
    const float* b_hh  = (const float*)d_in[4];
    const float* W_lin = (const float*)d_in[5];
    const float* b_lin = (const float*)d_in[6];
    const float* h0    = (const float*)d_in[7];
    const float* c0    = (const float*)d_in[8];
    float* out = (float*)d_out;

    lstm_persistent_kernel<<<1, NTHREADS, 0, stream>>>(
        x, W_ih, W_hh, b_ih, b_hh, W_lin, b_lin, h0, c0, out);
}
